// CoDeformableDetrTransformerEncoder_65051574665315
// MI455X (gfx1250) — matrix operand order repaired and checked by reading, not compiled, not run on hardware
//
#include <hip/hip_runtime.h>
#include <stdint.h>

// ---------------- static problem config ----------------
#define BATCH   2
#define NKEYS   13294
#define MTOK    (BATCH * NKEYS)   // 26588 tokens total
#define MR      26624             // MTOK rounded up to 256 (GEMM grid granularity)
#define EDIM    256
#define FDIM    1024
#define ZW      640               // fused proj width: v(256) | off(256) | aw(128)
#define NLAYERS 6

typedef __attribute__((ext_vector_type(16))) __bf16 v16bf;
typedef __attribute__((ext_vector_type(8)))  float  v8f;

__device__ __forceinline__ unsigned short f2bf(float f) {
    unsigned int u = __float_as_uint(f);
    u += 0x7FFFu + ((u >> 16) & 1u);          // round-to-nearest-even
    return (unsigned short)(u >> 16);
}

// generic LDS pointer -> 32-bit LDS byte offset (aperture lives in addr[63:32])
__device__ __forceinline__ unsigned lds_off_u32(const void* p) {
    return (unsigned)(uintptr_t)p;
}

// ---------------- weight packing into WMMA B-tile layout ----------------
// B tile (32x16 bf16) lane layout (wave32): half=lane>>4, n=lane&15.
// element e (0..15) of lane -> B[kt*32 + half*16 + e][nt*16 + n].
// Packed: P[((kt*NT16 + nt)*32 + lane)*16 + e]

__global__ __launch_bounds__(256) void pack_b_kernel(
    const float* __restrict__ W, unsigned short* __restrict__ P, int K, int NOUT)
{
    int idx = blockIdx.x * 256 + threadIdx.x;
    if (idx >= K * NOUT) return;
    int e    = idx & 15;
    int lane = (idx >> 4) & 31;
    int tile = idx >> 9;
    int nt16 = NOUT >> 4;
    int nt = tile % nt16, kt = tile / nt16;
    int k = kt * 32 + (lane >> 4) * 16 + e;
    int n = nt * 16 + (lane & 15);
    P[idx] = f2bf(W[(size_t)k * NOUT + n]);
}

// packed fused weight: columns [0,256)=Wv, [256,512)=Woff, [512,640)=Waw
__global__ __launch_bounds__(256) void pack_bz_kernel(
    const float* __restrict__ Wv, const float* __restrict__ Woff,
    const float* __restrict__ Waw, unsigned short* __restrict__ P)
{
    const int K = EDIM, NOUT = ZW;
    int idx = blockIdx.x * 256 + threadIdx.x;
    if (idx >= K * NOUT) return;
    int e    = idx & 15;
    int lane = (idx >> 4) & 31;
    int tile = idx >> 9;
    int nt16 = NOUT >> 4;
    int nt = tile % nt16, kt = tile / nt16;
    int k = kt * 32 + (lane >> 4) * 16 + e;
    int n = nt * 16 + (lane & 15);
    float w;
    if (n < 256)      w = Wv [(size_t)k * 256 + n];
    else if (n < 512) w = Woff[(size_t)k * 256 + (n - 256)];
    else              w = Waw[(size_t)k * 128 + (n - 512)];
    P[idx] = f2bf(w);
}

__global__ void pack_bias_kernel(const float* __restrict__ bv, const float* __restrict__ boff,
                                 const float* __restrict__ baw, float* __restrict__ bz)
{
    int i = blockIdx.x * 256 + threadIdx.x;
    if (i < 256)      bz[i] = bv[i];
    else if (i < 512) bz[i] = boff[i - 256];
    else if (i < 640) bz[i] = baw[i - 512];
}

__global__ void init_x_kernel(const float* __restrict__ src, float* __restrict__ x,
                              unsigned short* __restrict__ xb, int n)
{
    int i = blockIdx.x * 256 + threadIdx.x;
    if (i < n) { float v = src[i]; x[i] = v; xb[i] = f2bf(v); }
}

// ---------------- bf16 WMMA GEMM: Out[M,NOUT] = A[M,KDIM] * Bpacked + bias ----------------
// 4 waves / block, each wave computes a 64x32 tile (4 M-tiles x 2 N-tiles, 8 f32 accums).
// B tiles for the block (2 KB / K-step, shared by all 4 waves) are staged into LDS with
// GLOBAL_LOAD_ASYNC_TO_LDS_B128 (ASYNCcnt), double-buffered; A fragments are prefetched
// one K-step ahead into a shadow register set so global loads overlap the WMMA bursts.
template<int KDIM, int NOUT, bool RELU_BF16>
__global__ __launch_bounds__(128) void gemm_bf16(
    const unsigned short* __restrict__ A,   // M x KDIM row-major bf16
    const unsigned short* __restrict__ Bp,  // packed weight tiles
    const float* __restrict__ bias,         // NOUT
    void* __restrict__ Out, int M)
{
    __shared__ __align__(32) unsigned short bstage[2][1024];   // 2 x 2KB double buffer

    const int tid  = threadIdx.x;
    const int lane = tid & 31;
    const int wave = tid >> 5;
    const int mrow = (blockIdx.x * 4 + wave) * 64;
    const int ncol = blockIdx.y * 32;
    const int half = lane >> 4;
    const int l16  = lane & 15;
    const int NT16 = NOUT / 16;
    const int nt0  = ncol / 16;
    constexpr int KT = KDIM / 32;

    // this block's pair of adjacent B tiles is contiguous: 2KB per K-step
    const char*    bsrc      = (const char*)(Bp + (size_t)nt0 * 512) + (size_t)tid * 16;
    const unsigned stride_kt = (unsigned)NT16 * 1024u;   // bytes between K-steps
    unsigned lds_w[2];
    lds_w[0] = lds_off_u32(&bstage[0][(unsigned)tid * 8]);
    lds_w[1] = lds_off_u32(&bstage[1][(unsigned)tid * 8]);

    union ABu { uint4 u[2]; v16bf v; };

    // issue async copy of B tiles for K-step kt into stage buf (one b128 per thread)
    auto async_b = [&](int kt, int buf) {
        const char* g = bsrc + (size_t)kt * stride_kt;
        asm volatile("global_load_async_to_lds_b128 %0, %1, off"
                     :: "v"(lds_w[buf]), "v"(g) : "memory");
    };
    // load the 4 A fragments of K-step kt (two 16B runs per fragment)
    auto load_a = [&](ABu* dst, int kt) {
#pragma unroll
        for (int mi = 0; mi < 4; ++mi) {
            int m = mrow + mi * 16 + l16;
            if (m >= M) m = M - 1;                       // clamp OOB loads
            const unsigned short* ar = A + (size_t)m * KDIM + kt * 32 + half * 16;
            dst[mi].u[0] = *reinterpret_cast<const uint4*>(ar);
            dst[mi].u[1] = *reinterpret_cast<const uint4*>(ar + 8);
        }
    };

    v8f acc[4][2];
#pragma unroll
    for (int mi = 0; mi < 4; ++mi)
#pragma unroll
        for (int ni = 0; ni < 2; ++ni)
            acc[mi][ni] = (v8f){0.f,0.f,0.f,0.f,0.f,0.f,0.f,0.f};

    ABu a_cur[4], a_nxt[4];
    async_b(0, 0);
    load_a(a_cur, 0);

    for (int kt = 0; kt < KT; ++kt) {
        asm volatile("s_wait_asynccnt 0x0" ::: "memory");  // our stage-(kt&1) copy landed
        __syncthreads();                                   // visible to all waves; frees other stage
        const bool more = (kt + 1 < KT);
        if (more) {
            async_b(kt + 1, (kt + 1) & 1);
            load_a(a_nxt, kt + 1);
        }
        v16bf bfrag[2];
        const v16bf* bl = reinterpret_cast<const v16bf*>(&bstage[kt & 1][0]);
#pragma unroll
        for (int ni = 0; ni < 2; ++ni)
            bfrag[ni] = bl[ni * 32 + lane];
#pragma unroll
        for (int mi = 0; mi < 4; ++mi)
#pragma unroll
            for (int ni = 0; ni < 2; ++ni)
                acc[mi][ni] = __builtin_amdgcn_wmma_f32_16x16x32_bf16(
                    false, a_cur[mi].v, false, bfrag[ni], (short)0, acc[mi][ni], false, false);
        if (more) {
#pragma unroll
            for (int mi = 0; mi < 4; ++mi) a_cur[mi] = a_nxt[mi];
        }
    }

#pragma unroll
    for (int ni = 0; ni < 2; ++ni) {
        const int n  = ncol + ni * 16 + l16;
        const float bvv = bias ? bias[n] : 0.f;
#pragma unroll
        for (int mi = 0; mi < 4; ++mi) {
#pragma unroll
            for (int r = 0; r < 8; ++r) {
                int m = mrow + mi * 16 + half * 8 + r;   // C/D layout: lanes16-31 => M+8
                if (m < M) {
                    float val = acc[mi][ni][r] + bvv;
                    if constexpr (RELU_BF16) {
                        val = fmaxf(val, 0.f);
                        ((unsigned short*)Out)[(size_t)m * NOUT + n] = f2bf(val);
                    } else {
                        ((float*)Out)[(size_t)m * NOUT + n] = val;
                    }
                }
            }
        }
    }
}

// ---------------- deformable attention sampling ----------------
// block = 1 token, 8 waves = 8 heads, lane = channel d (0..31).
__global__ __launch_bounds__(256) void deform_kernel(
    const float* __restrict__ z,      // M x 640: v | off | aw_logits
    const float* __restrict__ ref,    // (B,N,4,2)
    unsigned short* __restrict__ attnb)
{
    const int t    = blockIdx.x;             // token index over B*N
    const int h    = threadIdx.x >> 5;       // head
    const int lane = threadIdx.x & 31;       // channel
    const int b    = t / NKEYS;

    const int   WW[4] = {100, 50, 25, 13};
    const int   HH[4] = {100, 50, 25, 13};
    const int   ST[4] = {0, 10000, 12500, 13125};

    const float* zt = z + (size_t)t * ZW;

    // softmax over the 16 (L*P) attention logits of this head
    float lg[16];
    float mx = -1e30f;
#pragma unroll
    for (int i = 0; i < 16; ++i) { lg[i] = zt[512 + h * 16 + i]; mx = fmaxf(mx, lg[i]); }
    float s = 0.f;
#pragma unroll
    for (int i = 0; i < 16; ++i) { lg[i] = __expf(lg[i] - mx); s += lg[i]; }
    const float inv = 1.f / s;

    const float* vbase = z + (size_t)b * NKEYS * ZW + h * 32 + lane;
    float acc = 0.f;

#pragma unroll
    for (int l = 0; l < 4; ++l) {
        const int   wl = WW[l], hl = HH[l], st = ST[l];
        const float wid = (float)wl, hgt = (float)hl;
        const float rx = ref[((size_t)t * 4 + l) * 2 + 0];
        const float ry = ref[((size_t)t * 4 + l) * 2 + 1];
#pragma unroll
        for (int p = 0; p < 4; ++p) {
            const float ox = zt[256 + ((h * 4 + l) * 4 + p) * 2 + 0];
            const float oy = zt[256 + ((h * 4 + l) * 4 + p) * 2 + 1];
            // loc = ref + off/normalizer ; X = loc_x*w - 0.5 = rx*w + ox - 0.5
            const float X = rx * wid + ox - 0.5f;
            const float Y = ry * hgt + oy - 0.5f;
            const float x0f = floorf(X), y0f = floorf(Y);
            const float lx = X - x0f, ly = Y - y0f;
            const int x0 = (int)x0f, y0 = (int)y0f;
            const float wgt = lg[l * 4 + p] * inv;
            const float w00 = (1.f - lx) * (1.f - ly) * wgt;
            const float w10 = lx * (1.f - ly) * wgt;
            const float w01 = (1.f - lx) * ly * wgt;
            const float w11 = lx * ly * wgt;
            if (x0 >= 0     && x0 < wl     && y0 >= 0     && y0 < hl)
                acc += w00 * vbase[(size_t)(st + y0 * wl + x0) * ZW];
            if (x0 + 1 >= 0 && x0 + 1 < wl && y0 >= 0     && y0 < hl)
                acc += w10 * vbase[(size_t)(st + y0 * wl + x0 + 1) * ZW];
            if (x0 >= 0     && x0 < wl     && y0 + 1 >= 0 && y0 + 1 < hl)
                acc += w01 * vbase[(size_t)(st + (y0 + 1) * wl + x0) * ZW];
            if (x0 + 1 >= 0 && x0 + 1 < wl && y0 + 1 >= 0 && y0 + 1 < hl)
                acc += w11 * vbase[(size_t)(st + (y0 + 1) * wl + x0 + 1) * ZW];
        }
    }
    attnb[(size_t)t * EDIM + h * 32 + lane] = f2bf(acc);
}

// ---------------- residual + LayerNorm (wave per token, 8 ch per lane) ----------------
__global__ __launch_bounds__(256) void ln_kernel(
    const float* __restrict__ res, const float* __restrict__ add,
    const float* __restrict__ g, const float* __restrict__ be,
    float* __restrict__ outf, unsigned short* __restrict__ outb, int M)
{
    const int t = blockIdx.x * 8 + (threadIdx.x >> 5);
    if (t >= M) return;
    const int lane = threadIdx.x & 31;
    float s[8]; float sum = 0.f, sq = 0.f;
#pragma unroll
    for (int i = 0; i < 8; ++i) {
        int c = i * 32 + lane;
        float v = res[(size_t)t * EDIM + c] + add[(size_t)t * EDIM + c];
        s[i] = v; sum += v; sq += v * v;
    }
#pragma unroll
    for (int o = 16; o > 0; o >>= 1) {
        sum += __shfl_xor(sum, o, 32);
        sq  += __shfl_xor(sq,  o, 32);
    }
    const float mean = sum * (1.f / EDIM);
    const float var  = sq  * (1.f / EDIM) - mean * mean;
    const float rstd = rsqrtf(var + 1e-5f);
#pragma unroll
    for (int i = 0; i < 8; ++i) {
        int c = i * 32 + lane;
        float v = (s[i] - mean) * rstd * g[c] + be[c];
        outf[(size_t)t * EDIM + c] = v;
        outb[(size_t)t * EDIM + c] = f2bf(v);
    }
}

// ---------------- host orchestration ----------------
extern "C" void kernel_launch(void* const* d_in, const int* in_sizes, int n_in,
                              void* d_out, int out_size, void* d_ws, size_t ws_size,
                              hipStream_t stream) {
    const float* src  = (const float*)d_in[0];
    const float* ref  = (const float*)d_in[1];
    const float* Woff = (const float*)d_in[4];
    const float* boff = (const float*)d_in[5];
    const float* Waw  = (const float*)d_in[6];
    const float* baw  = (const float*)d_in[7];
    const float* Wv   = (const float*)d_in[8];
    const float* bv   = (const float*)d_in[9];
    const float* Wo   = (const float*)d_in[10];
    const float* bo   = (const float*)d_in[11];
    const float* ln1g = (const float*)d_in[12];
    const float* ln1b = (const float*)d_in[13];
    const float* Wf1  = (const float*)d_in[14];
    const float* bf1  = (const float*)d_in[15];
    const float* Wf2  = (const float*)d_in[16];
    const float* bf2  = (const float*)d_in[17];
    const float* ln2g = (const float*)d_in[18];
    const float* ln2b = (const float*)d_in[19];

    float* x = (float*)d_out;   // current activations live in d_out (M x 256 fp32)

    // workspace carve-up (~165 MB)
    char* ws = (char*)d_ws;
    size_t off = 0;
    auto carve = [&](size_t bytes) { char* p = ws + off; off += (bytes + 255) & ~(size_t)255; return p; };
    unsigned short* xb    = (unsigned short*)carve((size_t)MR * EDIM * 2);
    float*          zbuf  = (float*)         carve((size_t)MR * ZW * 4);   // also reused as hb
    unsigned short* hb    = (unsigned short*)zbuf;                         // M x 1024 bf16 (fits in z)
    unsigned short* attnb = (unsigned short*)carve((size_t)MR * EDIM * 2);
    float*          go    = (float*)         carve((size_t)MR * EDIM * 4);
    float*          x1    = (float*)         carve((size_t)MR * EDIM * 4);
    unsigned short* x1b   = (unsigned short*)carve((size_t)MR * EDIM * 2);
    float*          bz    = (float*)         carve((size_t)ZW * 4);
    unsigned short* Wz_p  = (unsigned short*)carve((size_t)EDIM * ZW * 2);
    unsigned short* Wo_p  = (unsigned short*)carve((size_t)EDIM * EDIM * 2);
    unsigned short* Wf1_p = (unsigned short*)carve((size_t)EDIM * FDIM * 2);
    unsigned short* Wf2_p = (unsigned short*)carve((size_t)FDIM * EDIM * 2);
    (void)ws_size; (void)in_sizes; (void)n_in; (void)out_size;

    // pack weights (cheap; runs every call for determinism)
    pack_bz_kernel<<<(EDIM * ZW + 255) / 256, 256, 0, stream>>>(Wv, Woff, Waw, Wz_p);
    pack_bias_kernel<<<3, 256, 0, stream>>>(bv, boff, baw, bz);
    pack_b_kernel<<<(EDIM * EDIM + 255) / 256, 256, 0, stream>>>(Wo,  Wo_p,  EDIM, EDIM);
    pack_b_kernel<<<(EDIM * FDIM + 255) / 256, 256, 0, stream>>>(Wf1, Wf1_p, EDIM, FDIM);
    pack_b_kernel<<<(FDIM * EDIM + 255) / 256, 256, 0, stream>>>(Wf2, Wf2_p, FDIM, EDIM);
    init_x_kernel<<<(MTOK * EDIM + 255) / 256, 256, 0, stream>>>(src, x, xb, MTOK * EDIM);

    const dim3 gblk(128);
    const int  gx = MR / 256;   // 4 waves x 64 rows per block

    for (int layer = 0; layer < NLAYERS; ++layer) {
        // fused v/off/aw projection: z = x @ [Wv|Woff|Waw] + bias
        gemm_bf16<EDIM, ZW, false><<<dim3(gx, ZW / 32), gblk, 0, stream>>>(xb, Wz_p, bz, zbuf, MTOK);
        // deformable sampling -> attnb (bf16)
        deform_kernel<<<MTOK, 256, 0, stream>>>(zbuf, ref, attnb);
        // output projection
        gemm_bf16<EDIM, EDIM, false><<<dim3(gx, EDIM / 32), gblk, 0, stream>>>(attnb, Wo_p, bo, go, MTOK);
        // x1 = LN(x + attn)
        ln_kernel<<<(MTOK + 7) / 8, 256, 0, stream>>>(x, go, ln1g, ln1b, x1, x1b, MTOK);
        // FFN up (fused relu, bf16 store into hb == zbuf region)
        gemm_bf16<EDIM, FDIM, true><<<dim3(gx, FDIM / 32), gblk, 0, stream>>>(x1b, Wf1_p, bf1, hb, MTOK);
        // FFN down
        gemm_bf16<FDIM, EDIM, false><<<dim3(gx, EDIM / 32), gblk, 0, stream>>>(hb, Wf2_p, bf2, go, MTOK);
        // x = LN(x1 + ffn)  (writes fp32 back to d_out + bf16 for next layer's GEMM1)
        ln_kernel<<<(MTOK + 7) / 8, 256, 0, stream>>>(x1, go, ln2g, ln2b, x, xb, MTOK);
    }
}
